// CrossAttentionNodesBlockv2_11570641895552
// MI455X (gfx1250) — matrix operand order repaired and checked by reading, not compiled, not run on hardware
//
#include <hip/hip_runtime.h>
#include <hip/hip_bf16.h>

// ---------------------------------------------------------------------------
// CDNA5 / gfx1250 cross-attention block (TCR <-> pMHC), f16 WMMA everywhere.
// Round 2: GEMM reworked to BM=128/BN=128/BK=32 with double-buffered LDS and
// register-staged software pipelining (global loads overlap WMMA).
// ---------------------------------------------------------------------------

typedef __attribute__((ext_vector_type(16))) _Float16 v16h;
typedef __attribute__((ext_vector_type(8)))  float    v8f;
typedef _Float16 half_t;

static constexpr int DIM   = 512;
static constexpr int HEADS = 8;
static constexpr int HD    = 64;     // DIM / HEADS
static constexpr int NB    = 32;     // batch
static constexpr int LT    = 256;
static constexpr int LP    = 192;

union FragH { v16h v; unsigned u[8]; };

// --- fragment loaders (wave32 WMMA 16x16x32 f16 layouts, ISA 7.12.2) -------
// A (16x32 f16): lane m = l&15; lanes<16 hold K 0-7 & 16-23, lanes>=16 add 8.
__device__ __forceinline__ void load_fragA(FragH& f, const half_t* base,
                                           int strideHalves, int m, int hi) {
  const unsigned* row = (const unsigned*)(base + (size_t)m * strideHalves);
#pragma unroll
  for (int v = 0; v < 8; ++v) {
    int kp = (v < 4 ? 2 * v : 2 * v + 8) + (hi ? 8 : 0);  // even half index
    f.u[v] = row[kp >> 1];
  }
}
// B (32x16 f16) from an LDS tile stored [n][k]: lane n = l&15; lanes<16 hold
// K 0-7 & 16-23, lanes>=16 add 8 (same order as A).
__device__ __forceinline__ void load_fragB(FragH& f, const half_t* base,
                                           int strideHalves, int n, int hi,
                                           int kBaseUints) {
  const unsigned* row = (const unsigned*)(base + (size_t)n * strideHalves);
  int koff = kBaseUints + (hi ? 4 : 0);
#pragma unroll
  for (int v = 0; v < 8; ++v) f.u[v] = row[koff + (v < 4 ? v : v + 4)];
}

__device__ __forceinline__ v8f wmma_f16(const FragH& a, const FragH& b, v8f c) {
  return __builtin_amdgcn_wmma_f32_16x16x32_f16(false, a.v, false, b.v,
                                                (short)0, c, false, false);
}

// ---------------------------------------------------------------------------
// GEMM:  C[M,N] = A[M,K](f16,row-major) * W[K,N](f16,row-major) + bias
// BM=128, BN=128, BK=32. 256 threads = 8 waves (4 in M x 2 in N); each wave
// owns a 32x64 sub-tile = 2x4 WMMA tiles (8 accumulators).
// Double-buffered LDS + register staging: next tile's global loads issue
// before this tile's WMMAs; one barrier per K-step.
// ---------------------------------------------------------------------------
template <bool LEAKY, bool OUT16>
__global__ __launch_bounds__(256) void gemm_kernel(
    const half_t* __restrict__ A, const half_t* __restrict__ W,
    const float* __restrict__ bias, half_t* __restrict__ C16,
    float* __restrict__ C32, int M, int N, int K) {
  __shared__ half_t sA[2][128 * 40];  // stride 40 halves = 80B (16B rows)
  __shared__ half_t sB[2][128 * 40];  // transposed W tile: [n][k]

  const int tid  = threadIdx.x;
  const int lane = tid & 31;
  const int wave = tid >> 5;
  const int hi   = (lane >> 4) & 1;
  const int l15  = lane & 15;
  const int wm   = wave >> 1;  // 0..3 (M)
  const int wn   = wave & 1;   // 0..1 (N)
  const int m0   = blockIdx.y * 128;
  const int n0   = blockIdx.x * 128;

  // per-thread load slots
  const int ar = tid >> 1, ac = (tid & 1) * 16;      // A: 128 rows x 32 halves
  const int bk = tid >> 3, bn8 = (tid & 7) * 16;     // B: 32 k x 128 n
  const half_t* Asrc = A + (size_t)(m0 + ar) * K + ac;
  const half_t* Wsrc = W + (size_t)bk * N + n0 + bn8;

  v8f acc[2][4];
  {
    v8f z = {};
#pragma unroll
    for (int mt = 0; mt < 2; ++mt)
#pragma unroll
      for (int nt = 0; nt < 4; ++nt) acc[mt][nt] = z;
  }

  // ---- prologue: stage K-step 0 and commit to buffer 0 ----
  float4 ra0 = *(const float4*)(Asrc);
  float4 ra1 = *(const float4*)(Asrc + 8);
  float4 rb0 = *(const float4*)(Wsrc);
  float4 rb1 = *(const float4*)(Wsrc + 8);
  *(float4*)(&sA[0][ar * 40 + ac])     = ra0;
  *(float4*)(&sA[0][ar * 40 + ac + 8]) = ra1;
  {
    const half_t* t0 = (const half_t*)&rb0;
    const half_t* t1 = (const half_t*)&rb1;
#pragma unroll
    for (int j = 0; j < 8; ++j) sB[0][(bn8 + j) * 40 + bk] = t0[j];
#pragma unroll
    for (int j = 0; j < 8; ++j) sB[0][(bn8 + 8 + j) * 40 + bk] = t1[j];
  }
  __syncthreads();

  const int KI = K >> 5;  // K / 32
  for (int ki = 0; ki < KI; ++ki) {
    const int cur = ki & 1, nxt = cur ^ 1;
    const bool has_next = (ki + 1 < KI);

    // stage next tile into registers (overlaps with WMMA below)
    if (has_next) {
      const half_t* a = Asrc + (ki + 1) * 32;
      const half_t* w = Wsrc + (size_t)(ki + 1) * 32 * N;
      ra0 = *(const float4*)a;
      ra1 = *(const float4*)(a + 8);
      rb0 = *(const float4*)w;
      rb1 = *(const float4*)(w + 8);
      if (ki + 2 < KI) {
        __builtin_prefetch(Asrc + (ki + 2) * 32, 0, 3);  // global_prefetch_b8
      }
    }

    // compute current tile: 2 A-frags x 4 B-frags -> 8 WMMAs
    FragH fa0, fa1, fb[4];
    load_fragA(fa0, sA[cur], 40, wm * 32 + l15, hi);
    load_fragA(fa1, sA[cur], 40, wm * 32 + 16 + l15, hi);
#pragma unroll
    for (int nt = 0; nt < 4; ++nt)
      load_fragB(fb[nt], sB[cur], 40, wn * 64 + nt * 16 + l15, hi, 0);
#pragma unroll
    for (int nt = 0; nt < 4; ++nt) {
      acc[0][nt] = wmma_f16(fa0, fb[nt], acc[0][nt]);
      acc[1][nt] = wmma_f16(fa1, fb[nt], acc[1][nt]);
    }

    // commit staged tile to the other buffer
    if (has_next) {
      *(float4*)(&sA[nxt][ar * 40 + ac])     = ra0;
      *(float4*)(&sA[nxt][ar * 40 + ac + 8]) = ra1;
      const half_t* t0 = (const half_t*)&rb0;
      const half_t* t1 = (const half_t*)&rb1;
#pragma unroll
      for (int j = 0; j < 8; ++j) sB[nxt][(bn8 + j) * 40 + bk] = t0[j];
#pragma unroll
      for (int j = 0; j < 8; ++j) sB[nxt][(bn8 + 8 + j) * 40 + bk] = t1[j];
    }
    __syncthreads();
  }

  // epilogue: C/D layout -> lane col = l&15, VGPR r -> row r + 8*hi
#pragma unroll
  for (int mt = 0; mt < 2; ++mt) {
    const int mb = m0 + wm * 32 + mt * 16 + hi * 8;
#pragma unroll
    for (int nt = 0; nt < 4; ++nt) {
      const int gc = n0 + wn * 64 + nt * 16 + l15;
      const float bs = bias[gc];
#pragma unroll
      for (int rr = 0; rr < 8; ++rr) {
        float v = acc[mt][nt][rr] + bs;
        if (LEAKY) v = v >= 0.f ? v : 0.01f * v;
        if (OUT16)
          C16[(size_t)(mb + rr) * N + gc] = (half_t)v;
        else
          C32[(size_t)(mb + rr) * N + gc] = v;
      }
    }
  }
}

// ---------------------------------------------------------------------------
// Attention kernel: one block per (b, head, 32-query tile). 256 threads.
// scores = Q K^T / 8 (+mask) -> softmax -> head-avg atomicAdd -> ctx = P V.
// ---------------------------------------------------------------------------
__global__ __launch_bounds__(256) void attn_kernel(
    const half_t* __restrict__ Q, const half_t* __restrict__ Km,
    const half_t* __restrict__ V, const unsigned char* __restrict__ kmask,
    int masked_when_zero, half_t* __restrict__ ctx,
    float* __restrict__ attn_avg, int Lq, int Lk) {
  __shared__ float  sS[32 * 256];   // 32KB scores (fp32)
  __shared__ half_t sP[32 * 264];   // 16.5KB probs (f16), padded stride
  __shared__ half_t sK[64 * 72];    // 9KB key chunk (stride 72 -> 16B rows)
  __shared__ half_t sVt[64 * 36];   // 4.5KB transposed V chunk [n][k]

  const int tid  = threadIdx.x;
  const int lane = tid & 31;
  const int wave = tid >> 5;
  const int hi   = (lane >> 4) & 1;
  const int l15  = lane & 15;
  const int b    = blockIdx.y / HEADS;
  const int h    = blockIdx.y % HEADS;
  const int qbase = blockIdx.x * 32;
  const int wq   = wave >> 2;  // 0..1  (query 16-tile)
  const int wn   = wave & 3;   // 0..3  (key / HD 16-tile)

  // preload Q A-fragments for both 32-wide K steps of HD=64
  FragH aq[2];
  {
    int m = qbase + wq * 16 + l15;
    const unsigned* qrow =
        (const unsigned*)(Q + ((size_t)(b * Lq + m)) * DIM + h * HD);
#pragma unroll
    for (int ks = 0; ks < 2; ++ks)
#pragma unroll
      for (int v = 0; v < 8; ++v) {
        int kp = (v < 4 ? 2 * v : 2 * v + 8) + (hi ? 8 : 0) + ks * 32;
        aq[ks].u[v] = qrow[kp >> 1];
      }
  }

  // ---------------- scores ----------------
  for (int k0 = 0; k0 < Lk; k0 += 64) {
    __syncthreads();
    {  // load 64 key rows x 64 head-features
      int r = tid >> 2, c = (tid & 3) * 16;
      const half_t* src = Km + ((size_t)(b * Lk + k0 + r)) * DIM + h * HD + c;
      float4 t0 = *(const float4*)src;
      float4 t1 = *(const float4*)(src + 8);
      *(float4*)(sK + r * 72 + c)     = t0;
      *(float4*)(sK + r * 72 + c + 8) = t1;
    }
    __syncthreads();
    v8f acc = {};
#pragma unroll
    for (int ks = 0; ks < 2; ++ks) {
      FragH bk;
      load_fragB(bk, sK, 72, wn * 16 + l15, hi, ks * 16);
      acc = wmma_f16(aq[ks], bk, acc);
    }
    {  // scale + key mask, write to score buffer
      int key = k0 + wn * 16 + l15;
      unsigned char mb = kmask[b * Lk + key];
      bool masked = masked_when_zero ? (mb == 0) : (mb != 0);
#pragma unroll
      for (int rr = 0; rr < 8; ++rr) {
        int qr = wq * 16 + rr + hi * 8;
        sS[qr * 256 + key] = masked ? -1e30f : acc[rr] * 0.125f;
      }
    }
  }
  __syncthreads();

  // ---------------- softmax (one thread per query row) ----------------
  if (tid < 32) {
    float* row = sS + tid * 256;
    float mx = -1e38f;
    for (int k = 0; k < Lk; ++k) mx = fmaxf(mx, row[k]);
    float sum = 0.f;
    for (int k = 0; k < Lk; ++k) {
      float e = __expf(row[k] - mx);
      row[k] = e;
      sum += e;
    }
    float inv = 1.f / sum;
    for (int k = 0; k < Lk; ++k) row[k] *= inv;
  }
  __syncthreads();

  // head-averaged attention out (+= p/H) and f16 conversion
  for (int idx = tid; idx < 32 * Lk; idx += 256) {
    int qq = idx / Lk, kk = idx - qq * Lk;
    float p = sS[qq * 256 + kk];
    atomicAdd(&attn_avg[((size_t)(b * Lq + qbase + qq)) * Lk + kk],
              p * (1.0f / HEADS));
    sP[qq * 264 + kk] = (half_t)p;
  }
  __syncthreads();

  // ---------------- out = P @ V ----------------
  v8f oacc = {};
  for (int k0 = 0; k0 < Lk; k0 += 32) {
    __syncthreads();
    {  // transposed V chunk: sVt[n][k]
      int kloc = tid & 31;
      int n8 = (tid >> 5) * 8;
      const half_t* src =
          V + ((size_t)(b * Lk + k0 + kloc)) * DIM + h * HD + n8;
      float4 t = *(const float4*)src;
      const half_t* th = (const half_t*)&t;
#pragma unroll
      for (int j = 0; j < 8; ++j) sVt[(n8 + j) * 36 + kloc] = th[j];
    }
    __syncthreads();
    FragH ap, bv;
    {
      const unsigned* prow = (const unsigned*)(sP + (wq * 16 + l15) * 264);
#pragma unroll
      for (int v = 0; v < 8; ++v) {
        int kp = k0 + (v < 4 ? 2 * v : 2 * v + 8) + (hi ? 8 : 0);
        ap.u[v] = prow[kp >> 1];
      }
    }
    load_fragB(bv, sVt, 36, wn * 16 + l15, hi, 0);
    oacc = wmma_f16(ap, bv, oacc);
  }
  {
    int col = h * HD + wn * 16 + l15;
    int mrow = qbase + wq * 16 + hi * 8;
#pragma unroll
    for (int rr = 0; rr < 8; ++rr)
      ctx[((size_t)(b * Lq + mrow + rr)) * DIM + col] = (half_t)oacc[rr];
  }
}

// ---------------------------------------------------------------------------
// Residual + LayerNorm (fp32 + f16 outputs).  One block per row, DIM=512.
// ---------------------------------------------------------------------------
__global__ __launch_bounds__(256) void add_ln_kernel(
    const float* __restrict__ x, const float* __restrict__ r,
    const float* __restrict__ g, const float* __restrict__ bta,
    float* __restrict__ y32, half_t* __restrict__ y16) {
  const int row = blockIdx.x;
  const size_t base = (size_t)row * DIM;
  const int t = threadIdx.x;
  float v0 = x[base + t] + r[base + t];
  float v1 = x[base + t + 256] + r[base + t + 256];
  float s = v0 + v1, q = v0 * v0 + v1 * v1;
  __shared__ float redS[8], redQ[8];
#pragma unroll
  for (int o = 16; o > 0; o >>= 1) {
    s += __shfl_down(s, o, 32);
    q += __shfl_down(q, o, 32);
  }
  if ((t & 31) == 0) { redS[t >> 5] = s; redQ[t >> 5] = q; }
  __syncthreads();
  float ts = 0.f, tq = 0.f;
#pragma unroll
  for (int i = 0; i < 8; ++i) { ts += redS[i]; tq += redQ[i]; }
  float mean = ts * (1.0f / DIM);
  float var = tq * (1.0f / DIM) - mean * mean;
  float inv = rsqrtf(var + 1e-5f);
  float o0 = (v0 - mean) * inv * g[t] + bta[t];
  float o1 = (v1 - mean) * inv * g[t + 256] + bta[t + 256];
  y32[base + t] = o0;
  y32[base + t + 256] = o1;
  y16[base + t] = (half_t)o0;
  y16[base + t + 256] = (half_t)o1;
}

// Residual + LayerNorm + row masking, writes final fp32 output.
__global__ __launch_bounds__(256) void add_ln_mask_kernel(
    const float* __restrict__ x, const float* __restrict__ r,
    const float* __restrict__ g, const float* __restrict__ bta,
    const unsigned char* __restrict__ mA, int zeroA_if_nonzero,
    const unsigned char* __restrict__ mB, float* __restrict__ out) {
  const int row = blockIdx.x;
  const size_t base = (size_t)row * DIM;
  const int t = threadIdx.x;
  float v0 = x[base + t] + r[base + t];
  float v1 = x[base + t + 256] + r[base + t + 256];
  float s = v0 + v1, q = v0 * v0 + v1 * v1;
  __shared__ float redS[8], redQ[8];
#pragma unroll
  for (int o = 16; o > 0; o >>= 1) {
    s += __shfl_down(s, o, 32);
    q += __shfl_down(q, o, 32);
  }
  if ((t & 31) == 0) { redS[t >> 5] = s; redQ[t >> 5] = q; }
  __syncthreads();
  float ts = 0.f, tq = 0.f;
#pragma unroll
  for (int i = 0; i < 8; ++i) { ts += redS[i]; tq += redQ[i]; }
  float mean = ts * (1.0f / DIM);
  float var = tq * (1.0f / DIM) - mean * mean;
  float inv = rsqrtf(var + 1e-5f);
  bool zero = false;
  if (mA) {
    unsigned char a = mA[row];
    zero |= zeroA_if_nonzero ? (a != 0) : (a == 0);
  }
  if (mB) zero |= (mB[row] != 0);
  float o0 = zero ? 0.f : (v0 - mean) * inv * g[t] + bta[t];
  float o1 = zero ? 0.f : (v1 - mean) * inv * g[t + 256] + bta[t + 256];
  out[base + t] = o0;
  out[base + t + 256] = o1;
}

__global__ void cvt_kernel(const float* __restrict__ in,
                           half_t* __restrict__ out, size_t n) {
  size_t i = (size_t)blockIdx.x * 256 + threadIdx.x;
  if (i < n) out[i] = (half_t)in[i];
}

__global__ void zero_kernel(float* __restrict__ p, size_t n) {
  size_t i = (size_t)blockIdx.x * 256 + threadIdx.x;
  if (i < n) p[i] = 0.f;
}

// ---------------------------------------------------------------------------
// Host orchestration
// ---------------------------------------------------------------------------
struct Params {
  const float *tp_Wq, *tp_Wk, *tp_Wv, *tp_Wo, *tp_bq, *tp_bk, *tp_bv, *tp_bo;
  const float *pt_Wq, *pt_Wk, *pt_Wv, *pt_Wo, *pt_bq, *pt_bk, *pt_bv, *pt_bo;
  const float *tcr_ln1_g, *tcr_ln1_b, *tcr_ln2_g, *tcr_ln2_b;
  const float *tcr_W1, *tcr_b1, *tcr_W2, *tcr_b2;
  const float *pep_ln1_g, *pep_ln1_b, *pep_ln2_g, *pep_ln2_b;
  const float *pep_W1, *pep_b1, *pep_W2, *pep_b2;
};

static void launch_gemm(const half_t* A, const half_t* W, const float* bias,
                        half_t* C16, float* C32, int M, int N, int K,
                        bool leaky, bool out16, hipStream_t s) {
  dim3 g(N / 128, M / 128), blk(256);
  if (out16) {
    if (leaky)
      gemm_kernel<true, true><<<g, blk, 0, s>>>(A, W, bias, C16, C32, M, N, K);
    else
      gemm_kernel<false, true><<<g, blk, 0, s>>>(A, W, bias, C16, C32, M, N, K);
  } else {
    gemm_kernel<false, false><<<g, blk, 0, s>>>(A, W, bias, C16, C32, M, N, K);
  }
}

extern "C" void kernel_launch(void* const* d_in, const int* in_sizes, int n_in,
                              void* d_out, int out_size, void* d_ws,
                              size_t ws_size, hipStream_t stream) {
  const float* tcr_x  = (const float*)d_in[0];
  const float* pmhc_x = (const float*)d_in[1];
  const unsigned char* pep_mask = (const unsigned char*)d_in[2];
  const unsigned char* tcr_pad  = (const unsigned char*)d_in[3];
  const unsigned char* pmhc_pad = (const unsigned char*)d_in[4];

  // param ordering: detect insertion-order vs alphabetical tree-flatten.
  const float* f[32];
  for (int i = 0; i < 32; ++i) f[i] = (const float*)d_in[5 + i];
  Params P;
  if (in_sizes[5] == 512 * 1024) {  // alphabetical: pep_W1 first
    P.pep_W1 = f[0];  P.pep_W2 = f[1];  P.pep_b1 = f[2];  P.pep_b2 = f[3];
    P.pep_ln1_b = f[4]; P.pep_ln1_g = f[5]; P.pep_ln2_b = f[6]; P.pep_ln2_g = f[7];
    P.pt_Wk = f[8];  P.pt_Wo = f[9];  P.pt_Wq = f[10]; P.pt_Wv = f[11];
    P.pt_bk = f[12]; P.pt_bo = f[13]; P.pt_bq = f[14]; P.pt_bv = f[15];
    P.tcr_W1 = f[16]; P.tcr_W2 = f[17]; P.tcr_b1 = f[18]; P.tcr_b2 = f[19];
    P.tcr_ln1_b = f[20]; P.tcr_ln1_g = f[21]; P.tcr_ln2_b = f[22]; P.tcr_ln2_g = f[23];
    P.tp_Wk = f[24]; P.tp_Wo = f[25]; P.tp_Wq = f[26]; P.tp_Wv = f[27];
    P.tp_bk = f[28]; P.tp_bo = f[29]; P.tp_bq = f[30]; P.tp_bv = f[31];
  } else {  // insertion order
    P.tp_Wq = f[0];  P.tp_Wk = f[1];  P.tp_Wv = f[2];  P.tp_Wo = f[3];
    P.tp_bq = f[4];  P.tp_bk = f[5];  P.tp_bv = f[6];  P.tp_bo = f[7];
    P.pt_Wq = f[8];  P.pt_Wk = f[9];  P.pt_Wv = f[10]; P.pt_Wo = f[11];
    P.pt_bq = f[12]; P.pt_bk = f[13]; P.pt_bv = f[14]; P.pt_bo = f[15];
    P.tcr_ln1_g = f[16]; P.tcr_ln1_b = f[17]; P.tcr_ln2_g = f[18]; P.tcr_ln2_b = f[19];
    P.tcr_W1 = f[20]; P.tcr_b1 = f[21]; P.tcr_W2 = f[22]; P.tcr_b2 = f[23];
    P.pep_ln1_g = f[24]; P.pep_ln1_b = f[25]; P.pep_ln2_g = f[26]; P.pep_ln2_b = f[27];
    P.pep_W1 = f[28]; P.pep_b1 = f[29]; P.pep_W2 = f[30]; P.pep_b2 = f[31];
  }

  // output regions
  float* outp = (float*)d_out;
  float* out_tcr = outp;
  float* out_pep = out_tcr + (size_t)NB * LT * DIM;
  float* out_attn_tp = out_pep + (size_t)NB * LP * DIM;
  float* out_attn_pt = out_attn_tp + (size_t)NB * LT * LP;

  // workspace carve
  size_t off = 0;
  char* wsb = (char*)d_ws;
  auto carve_h = [&](size_t n) {
    half_t* p = (half_t*)(wsb + off);
    off = (off + n * sizeof(half_t) + 255) & ~(size_t)255;
    return p;
  };
  auto carve_f = [&](size_t n) {
    float* p = (float*)(wsb + off);
    off = (off + n * sizeof(float) + 255) & ~(size_t)255;
    return p;
  };
  const size_t nT = (size_t)NB * LT * DIM;  // 4,194,304
  const size_t nP = (size_t)NB * LP * DIM;  // 3,145,728
  half_t* tcr16  = carve_h(nT);
  half_t* pmhc16 = carve_h(nP);
  const size_t nW = (size_t)DIM * DIM, nW1 = (size_t)DIM * 2 * DIM;
  half_t* w_tp_q = carve_h(nW); half_t* w_tp_k = carve_h(nW);
  half_t* w_tp_v = carve_h(nW); half_t* w_tp_o = carve_h(nW);
  half_t* w_pt_q = carve_h(nW); half_t* w_pt_k = carve_h(nW);
  half_t* w_pt_v = carve_h(nW); half_t* w_pt_o = carve_h(nW);
  half_t* w_tcr_1 = carve_h(nW1); half_t* w_tcr_2 = carve_h(nW1);
  half_t* w_pep_1 = carve_h(nW1); half_t* w_pep_2 = carve_h(nW1);
  // reusable per-side scratch (sized for the larger side)
  half_t* Q16  = carve_h(nT);
  half_t* K16  = carve_h(nT);
  half_t* V16  = carve_h(nT);
  half_t* ctx16 = carve_h(nT);
  half_t* y116  = carve_h(nT);
  half_t* h116  = carve_h((size_t)NB * LT * 2 * DIM);
  float* attn32 = carve_f(nT);
  float* y132   = carve_f(nT);
  float* h232   = carve_f(nT);

  auto cvt = [&](const float* src, half_t* dst, size_t n) {
    cvt_kernel<<<dim3((unsigned)((n + 255) / 256)), dim3(256), 0, stream>>>(src, dst, n);
  };
  cvt(tcr_x, tcr16, nT);
  cvt(pmhc_x, pmhc16, nP);
  cvt(P.tp_Wq, w_tp_q, nW); cvt(P.tp_Wk, w_tp_k, nW);
  cvt(P.tp_Wv, w_tp_v, nW); cvt(P.tp_Wo, w_tp_o, nW);
  cvt(P.pt_Wq, w_pt_q, nW); cvt(P.pt_Wk, w_pt_k, nW);
  cvt(P.pt_Wv, w_pt_v, nW); cvt(P.pt_Wo, w_pt_o, nW);
  cvt(P.tcr_W1, w_tcr_1, nW1); cvt(P.tcr_W2, w_tcr_2, nW1);
  cvt(P.pep_W1, w_pep_1, nW1); cvt(P.pep_W2, w_pep_2, nW1);

  // zero both head-averaged attention outputs (accumulated via atomics)
  {
    size_t nz = (size_t)NB * LT * LP + (size_t)NB * LP * LT;
    zero_kernel<<<dim3((unsigned)((nz + 255) / 256)), dim3(256), 0, stream>>>(
        out_attn_tp, nz);
  }

  const int Mt = NB * LT;  // 8192
  const int Mp = NB * LP;  // 6144

  // ===================== side 1: TCR queries -> peptide keys ===============
  launch_gemm(tcr16,  w_tp_q, P.tp_bq, Q16, nullptr, Mt, DIM, DIM, false, true, stream);
  launch_gemm(pmhc16, w_tp_k, P.tp_bk, K16, nullptr, Mp, DIM, DIM, false, true, stream);
  launch_gemm(pmhc16, w_tp_v, P.tp_bv, V16, nullptr, Mp, DIM, DIM, false, true, stream);
  attn_kernel<<<dim3(LT / 32, NB * HEADS), dim3(256), 0, stream>>>(
      Q16, K16, V16, pep_mask, /*masked_when_zero=*/1, ctx16, out_attn_tp, LT, LP);
  launch_gemm(ctx16, w_tp_o, P.tp_bo, nullptr, attn32, Mt, DIM, DIM, false, false, stream);
  add_ln_kernel<<<dim3(Mt), dim3(256), 0, stream>>>(
      tcr_x, attn32, P.tcr_ln1_g, P.tcr_ln1_b, y132, y116);
  launch_gemm(y116, w_tcr_1, P.tcr_b1, h116, nullptr, Mt, 2 * DIM, DIM, true, true, stream);
  launch_gemm(h116, w_tcr_2, P.tcr_b2, nullptr, h232, Mt, DIM, 2 * DIM, false, false, stream);
  add_ln_mask_kernel<<<dim3(Mt), dim3(256), 0, stream>>>(
      y132, h232, P.tcr_ln2_g, P.tcr_ln2_b, tcr_pad, /*zero if nonzero*/ 1,
      nullptr, out_tcr);

  // ===================== side 2: pMHC queries -> TCR keys ==================
  launch_gemm(pmhc16, w_pt_q, P.pt_bq, Q16, nullptr, Mp, DIM, DIM, false, true, stream);
  launch_gemm(tcr16,  w_pt_k, P.pt_bk, K16, nullptr, Mt, DIM, DIM, false, true, stream);
  launch_gemm(tcr16,  w_pt_v, P.pt_bv, V16, nullptr, Mt, DIM, DIM, false, true, stream);
  attn_kernel<<<dim3(LP / 32, NB * HEADS), dim3(256), 0, stream>>>(
      Q16, K16, V16, tcr_pad, /*masked_when_zero=*/0, ctx16, out_attn_pt, LP, LT);
  launch_gemm(ctx16, w_pt_o, P.pt_bo, nullptr, attn32, Mp, DIM, DIM, false, false, stream);
  add_ln_kernel<<<dim3(Mp), dim3(256), 0, stream>>>(
      pmhc_x, attn32, P.pep_ln1_g, P.pep_ln1_b, y132, y116);
  launch_gemm(y116, w_pep_1, P.pep_b1, h116, nullptr, Mp, 2 * DIM, DIM, true, true, stream);
  launch_gemm(h116, w_pep_2, P.pep_b2, nullptr, h232, Mp, DIM, 2 * DIM, false, false, stream);
  add_ln_mask_kernel<<<dim3(Mp), dim3(256), 0, stream>>>(
      y132, h232, P.pep_ln2_g, P.pep_ln2_b, pep_mask, /*zero if zero*/ 0,
      pmhc_pad, out_pep);
}
